// SCAESuite_56530359550036
// MI455X (gfx1250) — compile-verified, matrix operand order checked by reading, not run on hardware
//
#include <hip/hip_runtime.h>
#include <hip/hip_bf16.h>

// ---------------- problem constants ----------------
constexpr int Bc = 4, Sc = 1024, Dc = 768, Fc = 8192, Uc = 3, Kt = 64;
constexpr int BS = Bc * Sc; // 4096

// ---------------- vector types ----------------
typedef __attribute__((ext_vector_type(16))) __bf16 v16bf;
typedef __attribute__((ext_vector_type(8)))  __bf16 v8bf;
typedef __attribute__((ext_vector_type(8)))  float  v8f;

#define LDS_STRIDE 40  // 32 bf16 payload + 8 pad (80B rows) to spread LDS banks

// ---------------- CDNA5 async global->LDS helpers ----------------
// 16B per lane, tracked with ASYNCcnt; bypasses VGPR staging entirely.
__device__ __forceinline__ void async_copy_b128(unsigned lds_off,
                                                const void* gptr) {
  asm volatile("global_load_async_to_lds_b128 %0, %1, off"
               :
               : "v"(lds_off), "v"((unsigned long long)gptr)
               : "memory");
}
__device__ __forceinline__ void wait_asynccnt0() {
#if __has_builtin(__builtin_amdgcn_s_wait_asynccnt)
  __builtin_amdgcn_s_wait_asynccnt(0);
#else
  asm volatile("s_wait_asynccnt 0x0" ::: "memory");
#endif
}
__device__ __forceinline__ void wait_asynccnt4() {
#if __has_builtin(__builtin_amdgcn_s_wait_asynccnt)
  __builtin_amdgcn_s_wait_asynccnt(4);
#else
  asm volatile("s_wait_asynccnt 0x4" ::: "memory");
#endif
}

// ---------------- f32 -> bf16 convert (grid-stride) ----------------
__global__ void convert_f32_bf16(const float* __restrict__ in,
                                 __bf16* __restrict__ out, size_t n) {
  size_t i = (size_t)blockIdx.x * blockDim.x + threadIdx.x;
  size_t stride = (size_t)gridDim.x * blockDim.x;
  for (; i < n; i += stride) out[i] = (__bf16)in[i];
}

// ---------------- tiled transpose: out[C][R] = in[R][C] ----------------
template <typename OutT>
__global__ void transpose_kernel(const float* __restrict__ in,
                                 OutT* __restrict__ out, int R, int C) {
  __shared__ float tile[32][33];
  int bx = blockIdx.x;  // tile along C
  int by = blockIdx.y;  // tile along R
  int tx = threadIdx.x, ty = threadIdx.y;  // 32 x 8
#pragma unroll
  for (int i = 0; i < 32; i += 8)
    tile[ty + i][tx] = in[(size_t)(by * 32 + ty + i) * C + bx * 32 + tx];
  __syncthreads();
#pragma unroll
  for (int i = 0; i < 32; i += 8)
    out[(size_t)(bx * 32 + ty + i) * R + by * 32 + tx] = (OutT)tile[tx][ty + i];
}

// ---------------- bias vectors: c1 = enc@(sum_u up_b_dec), c2 = enc_b - enc@b_dec ----
__global__ void bias_vec_kernel(const float* __restrict__ enc,
                                const float* __restrict__ enc_b,
                                const float* __restrict__ b_dec,
                                const float* __restrict__ up_b_dec,
                                float* __restrict__ c1, float* __restrict__ c2) {
  int f = blockIdx.x * blockDim.x + threadIdx.x;
  if (f >= Fc) return;
  const float* row = enc + (size_t)f * Dc;
  float s1 = 0.f, s2 = 0.f;
  for (int d = 0; d < Dc; ++d) {
    float w = row[d];
    s1 += w * b_dec[d];
    s2 += w * (up_b_dec[d] + up_b_dec[Dc + d] + up_b_dec[2 * Dc + d]);
  }
  c1[f] = s2;
  c2[f] = enc_b[f] - s1;
}

// ---------------- core bf16 WMMA GEMM (double-buffered, async-to-LDS) ----------------
// C[M,N] (f32) = A[M,K] (bf16, row-major) x B[N,K] (bf16, row-major, i.e. B^T)
// MODE 0: C = acc       MODE 1: C += acc
// MODE 2: Cb[M,N] (bf16) = Mask[M,N] ? acc : 0     (fused conn_mask + downconvert)
template <int MODE>
__global__ __launch_bounds__(256) void gemm_wmma_bf16(
    const __bf16* __restrict__ A, const __bf16* __restrict__ B,
    float* __restrict__ C, __bf16* __restrict__ Cb,
    const int* __restrict__ Mask, int M, int N, int Kdim) {
  __shared__ __align__(16) __bf16 ldsA[2][128 * LDS_STRIDE];
  __shared__ __align__(16) __bf16 ldsB[2][128 * LDS_STRIDE];

  const int tid   = threadIdx.x;
  const int lane  = tid & 31;
  const int wave  = tid >> 5;
  const int waveM = wave >> 1;  // 0..3 -> 32-row slab
  const int waveN = wave & 1;   // 0..1 -> 64-col slab
  const int tileN0 = blockIdx.x * 128;
  const int tileM0 = blockIdx.y * 128;

  const int fr = lane & 15;   // row (A) / col (B) within 16x16 tile
  const int g  = lane >> 4;   // K-half selector per ISA layout

  // this thread's two 16B staging chunks: (row, k-chunk) of the 128x32 tile
  const int r0 = tid >> 2;           // q = tid
  const int c0 = (tid & 3) * 8;
  const int r1 = (tid + 256) >> 2;   // q = tid + 256
  const int c1i = ((tid + 256) & 3) * 8;

  v8f zero = {0.f, 0.f, 0.f, 0.f, 0.f, 0.f, 0.f, 0.f};
  v8f acc[2][4];
#pragma unroll
  for (int i = 0; i < 2; ++i)
#pragma unroll
    for (int j = 0; j < 4; ++j) acc[i][j] = zero;

  // issue 4 async b128 copies for one K-slab into buffer `buf`
  auto issue_tile = [&](int kb, int buf) {
    async_copy_b128((unsigned)(size_t)&ldsA[buf][r0 * LDS_STRIDE + c0],
                    A + (size_t)(tileM0 + r0) * Kdim + kb + c0);
    async_copy_b128((unsigned)(size_t)&ldsB[buf][r0 * LDS_STRIDE + c0],
                    B + (size_t)(tileN0 + r0) * Kdim + kb + c0);
    async_copy_b128((unsigned)(size_t)&ldsA[buf][r1 * LDS_STRIDE + c1i],
                    A + (size_t)(tileM0 + r1) * Kdim + kb + c1i);
    async_copy_b128((unsigned)(size_t)&ldsB[buf][r1 * LDS_STRIDE + c1i],
                    B + (size_t)(tileN0 + r1) * Kdim + kb + c1i);
  };

  issue_tile(0, 0);
  int buf = 0;

  for (int kb = 0; kb < Kdim; kb += 32) {
    if (kb + 32 < Kdim) {
      issue_tile(kb + 32, buf ^ 1);  // prefetch next slab into other buffer
      wait_asynccnt4();              // only current slab (4 ops) must be done
    } else {
      wait_asynccnt0();
    }
    __syncthreads();  // all waves' async writes to buf now visible

    // fragments per ISA 16-bit A/B layout:
    //   lane L holds row/col L%16; VGPR0-3 hold K = g*8..g*8+7, VGPR4-7 hold K = 16+g*8..
    v16bf afrag[2], bfrag[4];
#pragma unroll
    for (int i = 0; i < 2; ++i) {
      const __bf16* p = &ldsA[buf][(waveM * 32 + i * 16 + fr) * LDS_STRIDE];
      v8bf lo = *(const v8bf*)(p + g * 8);
      v8bf hi = *(const v8bf*)(p + 16 + g * 8);
      afrag[i] = __builtin_shufflevector(lo, hi, 0, 1, 2, 3, 4, 5, 6, 7, 8, 9,
                                         10, 11, 12, 13, 14, 15);
    }
#pragma unroll
    for (int j = 0; j < 4; ++j) {
      const __bf16* p = &ldsB[buf][(waveN * 64 + j * 16 + fr) * LDS_STRIDE];
      v8bf lo = *(const v8bf*)(p + g * 8);
      v8bf hi = *(const v8bf*)(p + 16 + g * 8);
      bfrag[j] = __builtin_shufflevector(lo, hi, 0, 1, 2, 3, 4, 5, 6, 7, 8, 9,
                                         10, 11, 12, 13, 14, 15);
    }
#pragma unroll
    for (int i = 0; i < 2; ++i)
#pragma unroll
      for (int j = 0; j < 4; ++j)
        acc[i][j] = __builtin_amdgcn_wmma_f32_16x16x32_bf16(
            false, afrag[i], false, bfrag[j], (short)0, acc[i][j], false, false);
    __syncthreads();  // everyone done reading buf before it is overwritten
    buf ^= 1;
  }

  // epilogue; C/D layout: col = lane%16, VGPR e -> M = e + 8*(lane/16)
#pragma unroll
  for (int i = 0; i < 2; ++i) {
    int rowb = tileM0 + waveM * 32 + i * 16 + g * 8;
#pragma unroll
    for (int j = 0; j < 4; ++j) {
      int col = tileN0 + waveN * 64 + j * 16 + fr;
#pragma unroll
      for (int e = 0; e < 8; ++e) {
        size_t off = (size_t)(rowb + e) * N + col;
        float v = acc[i][j][e];
        if (MODE == 0) {
          C[off] = v;
        } else if (MODE == 1) {
          C[off] += v;
        } else {
          Cb[off] = Mask[off] ? (__bf16)v : (__bf16)0.0f;
        }
      }
    }
  }
}

// ---------------- per-row top-64 + ReLU + sparse reconstruction ----------------
__global__ __launch_bounds__(256) void topk_epilogue(
    const float* __restrict__ approx, const float* __restrict__ ln_scale,
    const float* __restrict__ c1, const float* __restrict__ c2,
    const float* __restrict__ ddT, const float* __restrict__ b_dec,
    float* __restrict__ recon_out) {
  __shared__ float vals[Fc];
  __shared__ float redv[256];
  __shared__ int   redi[256];
  __shared__ float topv[Kt];
  __shared__ int   topi[Kt];

  const int r = blockIdx.x;
  const int t = threadIdx.x;
  const float inv_s = 1.0f / ln_scale[r];
  const float* arow = approx + (size_t)r * Fc;

  for (int f = t; f < Fc; f += 256)
    vals[f] = (arow[f] + c1[f]) * inv_s + c2[f];
  __syncthreads();

  for (int k = 0; k < Kt; ++k) {
    float bv = -INFINITY;
    int bi = 0;
#pragma unroll
    for (int j = 0; j < Fc / 256; ++j) {
      int f = t + j * 256;  // stride-256 -> conflict-free banks
      float v = vals[f];
      if (v > bv) { bv = v; bi = f; }
    }
    redv[t] = bv;
    redi[t] = bi;
    __syncthreads();
    for (int s = 128; s > 0; s >>= 1) {
      if (t < s && redv[t + s] > redv[t]) {
        redv[t] = redv[t + s];
        redi[t] = redi[t + s];
      }
      __syncthreads();
    }
    if (t == 0) {
      int bidx = redi[0];
      topv[k] = redv[0] > 0.f ? redv[0] : 0.f;  // ReLU
      topi[k] = bidx;
      vals[bidx] = -INFINITY;
    }
    __syncthreads();
  }

  // recon[r, :] = b_dec + sum_k topv[k] * down_dec_w[:, topi[k]]  (ddT rows contiguous)
  float a0 = 0.f, a1 = 0.f, a2 = 0.f;
  for (int k = 0; k < Kt; ++k) {
    float v = topv[k];
    const float* row = ddT + (size_t)topi[k] * Dc;
    a0 += v * row[t];
    a1 += v * row[t + 256];
    a2 += v * row[t + 512];
  }
  float* out = recon_out + (size_t)r * Dc;
  out[t]       = a0 + b_dec[t];
  out[t + 256] = a1 + b_dec[t + 256];
  out[t + 512] = a2 + b_dec[t + 512];
}

// ---------------- host ----------------
extern "C" void kernel_launch(void* const* d_in, const int* in_sizes, int n_in,
                              void* d_out, int out_size, void* d_ws,
                              size_t ws_size, hipStream_t stream) {
  (void)in_sizes; (void)n_in; (void)out_size; (void)ws_size;

  const float* resid  = (const float*)d_in[0];   // [B,S,D]
  const float* ln     = (const float*)d_in[1];   // [B,S,1]
  const float* upf    = (const float*)d_in[2];   // [U,B,S,F]
  const float* fbuf   = (const float*)d_in[3];   // [B,S,F]
  const float* encw   = (const float*)d_in[4];   // [F,D]
  const float* encb   = (const float*)d_in[5];   // [F]
  const float* decw   = (const float*)d_in[6];   // [D,F]
  const float* bdec   = (const float*)d_in[7];   // [D]
  const float* updec  = (const float*)d_in[8];   // [U,D,F]
  const float* upbdec = (const float*)d_in[9];   // [U,D]
  const int*   mask   = (const int*)d_in[10];    // [U,F,F] bool as int

  // ---- workspace carve (256B aligned) ----
  char* w = (char*)d_ws;
  auto carve = [&](size_t bytes) {
    char* p = w;
    w += (bytes + 255) & ~(size_t)255;
    return (void*)p;
  };
  __bf16* encB   = (__bf16*)carve((size_t)Fc * Dc * 2);       // [F,D]
  __bf16* updecT = (__bf16*)carve((size_t)Uc * Fc * Dc * 2);  // [U,F,D]
  __bf16* residB = (__bf16*)carve((size_t)BS * Dc * 2);       // [BS,D]
  __bf16* upfB   = (__bf16*)carve((size_t)Uc * BS * Fc * 2);  // [U,BS,F]
  __bf16* vw     = (__bf16*)carve((size_t)Fc * Fc * 2);       // [F,F] reused per u
  float*  approx = (float*)carve((size_t)BS * Fc * 4);        // [BS,F]
  float*  ddT    = (float*)carve((size_t)Fc * Dc * 4);        // [F,D]
  float*  c1     = (float*)carve((size_t)Fc * 4);
  float*  c2     = (float*)carve((size_t)Fc * 4);

  // output[0]: the (unscattered) feature_buffer, faithfully copied through
  hipMemcpyAsync(d_out, fbuf, (size_t)BS * Fc * sizeof(float),
                 hipMemcpyDeviceToDevice, stream);
  float* recon = (float*)d_out + (size_t)BS * Fc;

  // ---- precompute / convert ----
  convert_f32_bf16<<<2048, 256, 0, stream>>>(encw, encB, (size_t)Fc * Dc);
  convert_f32_bf16<<<2048, 256, 0, stream>>>(resid, residB, (size_t)BS * Dc);
  convert_f32_bf16<<<16384, 256, 0, stream>>>(upf, upfB, (size_t)Uc * BS * Fc);

  dim3 tb(32, 8);
  dim3 tgDF(Fc / 32, Dc / 32);  // in [D,F] -> out [F,D]
  for (int u = 0; u < Uc; ++u)
    transpose_kernel<__bf16><<<tgDF, tb, 0, stream>>>(
        updec + (size_t)u * Dc * Fc, updecT + (size_t)u * Fc * Dc, Dc, Fc);
  transpose_kernel<float><<<tgDF, tb, 0, stream>>>(decw, ddT, Dc, Fc);

  bias_vec_kernel<<<Fc / 256, 256, 0, stream>>>(encw, encb, bdec, upbdec, c1, c2);

  // ---- GEMM 1: approx = resid x enc^T   [4096 x 8192, K=768] ----
  {
    dim3 g(Fc / 128, BS / 128);
    gemm_wmma_bf16<0><<<g, 256, 0, stream>>>(residB, encB, approx, nullptr,
                                             nullptr, BS, Fc, Dc);
  }

  // ---- per-upstream: vw = (enc x up_dec[u]) . mask  then approx += upf[u] x vw^T ----
  for (int u = 0; u < Uc; ++u) {
    dim3 g2(Fc / 128, Fc / 128);  // [8192 x 8192, K=768], masked bf16 out
    gemm_wmma_bf16<2><<<g2, 256, 0, stream>>>(
        encB, updecT + (size_t)u * Fc * Dc, nullptr, vw,
        mask + (size_t)u * Fc * Fc, Fc, Fc, Dc);
    dim3 g3(Fc / 128, BS / 128);  // [4096 x 8192, K=8192], accumulate
    gemm_wmma_bf16<1><<<g3, 256, 0, stream>>>(
        upfB + (size_t)u * BS * Fc, vw, approx, nullptr, nullptr, BS, Fc, Fc);
  }

  // ---- scale + bias + top-64 + ReLU + sparse decode ----
  topk_epilogue<<<BS, 256, 0, stream>>>(approx, ln, c1, c2, ddT, bdec, recon);
}
